// HierarchicalAttentionLayer_88484916232475
// MI455X (gfx1250) — matrix operand order repaired and checked by reading, not compiled, not run on hardware
//
#include <hip/hip_runtime.h>
#include <hip/hip_bf16.h>

#define B_SZ 2
#define L_SEQ 2048
#define D_MODEL 1024
#define N_HEADS 16
#define HEAD_DIM 64
#define M_ROWS (B_SZ * L_SEQ) /* 4096 */
#define WIN_HALF 128

typedef __attribute__((ext_vector_type(16))) _Float16 v16h;
typedef __attribute__((ext_vector_type(8)))  _Float16 v8h;
typedef __attribute__((ext_vector_type(8)))  float    v8f;

static __device__ __forceinline__ v16h ld_frag_2x8(const _Float16* p0, const _Float16* p1) {
    v8h lo = *(const v8h*)p0;
    v8h hi = *(const v8h*)p1;
    v16h r;
#pragma unroll
    for (int i = 0; i < 8; ++i) { r[i] = lo[i]; r[8 + i] = hi[i]; }
    return r;
}

static __device__ __forceinline__ v16h ld_frag_16(const _Float16* p) {
    return ld_frag_2x8(p, p + 8);
}

static __device__ __forceinline__ v8f wmma_f16(v16h a, v16h b, v8f c) {
    return __builtin_amdgcn_wmma_f32_16x16x32_f16(false, a, false, b, (short)0, c, false, false);
}

// ---------------------------------------------------------------------------
// Pack f32 weight W[K][N] into WMMA B-fragment order, grouped so one wave's
// four B fragments per k-step sit at one base pointer + immediate offsets:
//   Bp[(((nb*KT + kt)*4 + j)*32 + lane)*16 + e]
//     = W[kt*32 + (e>>3)*16 + (lane>>4)*8 + (e&7)][(nb*4 + j)*16 + (lane&15)]
// ---------------------------------------------------------------------------
__global__ void pack_w_kernel(const float* __restrict__ W, _Float16* __restrict__ Bp,
                              int K, int N) {
    const size_t idx = (size_t)blockIdx.x * 256 + threadIdx.x;
    if (idx >= (size_t)K * N) return;
    const int e    = (int)(idx & 15);
    const int lane = (int)((idx >> 4) & 31);
    const int j    = (int)((idx >> 9) & 3);
    const size_t t = idx >> 11;          // nb*KT + kt
    const int KT = K >> 5;
    const int kt = (int)(t % KT);
    const int nb = (int)(t / KT);
    const int n = ((nb << 2) + j) * 16 + (lane & 15);
    const int k = (kt << 5) + ((e >> 3) << 4) + ((lane >> 4) << 3) + (e & 7);
    Bp[idx] = (_Float16)W[(size_t)k * N + n];
}

// ---------------------------------------------------------------------------
// LayerNorm over D_MODEL, writes f16 activations. One block per token row.
// ---------------------------------------------------------------------------
__global__ void layernorm_kernel(const float* __restrict__ x,
                                 const float* __restrict__ g,
                                 const float* __restrict__ b,
                                 _Float16* __restrict__ out) {
    __shared__ float red[256];
    const int row = blockIdx.x;
    const int tid = threadIdx.x;
    const float* xr = x + (size_t)row * D_MODEL;
    float v[4];
    float s = 0.f;
#pragma unroll
    for (int i = 0; i < 4; ++i) { v[i] = xr[tid + (i << 8)]; s += v[i]; }
    red[tid] = s; __syncthreads();
    for (int off = 128; off > 0; off >>= 1) {
        if (tid < off) red[tid] += red[tid + off];
        __syncthreads();
    }
    const float mu = red[0] * (1.0f / D_MODEL);
    __syncthreads();
    float s2 = 0.f;
#pragma unroll
    for (int i = 0; i < 4; ++i) { float d = v[i] - mu; s2 += d * d; }
    red[tid] = s2; __syncthreads();
    for (int off = 128; off > 0; off >>= 1) {
        if (tid < off) red[tid] += red[tid + off];
        __syncthreads();
    }
    const float rs = rsqrtf(red[0] * (1.0f / D_MODEL) + 1e-5f);
    _Float16* orow = out + (size_t)row * D_MODEL;
#pragma unroll
    for (int i = 0; i < 4; ++i) {
        const int c = tid + (i << 8);
        orow[c] = (_Float16)((v[i] - mu) * rs * g[c] + b[c]);
    }
}

// ---------------------------------------------------------------------------
// GEMM fragment-set helpers (one k-step: 2 A fragments + 4 B fragments)
// ---------------------------------------------------------------------------
template <int K>
static __device__ __forceinline__ void load_set(const _Float16* ap, const _Float16* bp,
                                                v16h& a0, v16h& a1, v16h b[4]) {
    a0 = ld_frag_2x8(ap, ap + 16);
    a1 = ld_frag_2x8(ap + 16 * K, ap + 16 * K + 16);
#pragma unroll
    for (int j = 0; j < 4; ++j) b[j] = ld_frag_16(bp + j * 512);
}

static __device__ __forceinline__ void wmma_set(const v16h& a0, const v16h& a1,
                                                const v16h b[4], v8f acc[2][4]) {
#pragma unroll
    for (int j = 0; j < 4; ++j) {
        acc[0][j] = wmma_f16(a0, b[j], acc[0][j]);
        acc[1][j] = wmma_f16(a1, b[j], acc[1][j]);
    }
}

// ---------------------------------------------------------------------------
// WMMA GEMM: each wave computes a 32x64 tile of C = A(f16,row-major) @ Bp.
// K is a template constant (immediate load offsets); even/odd k-steps live in
// ping-pong register sets so loads for step kt+1 overlap the WMMAs of step kt.
// mode 0: QKV scatter -> q[b][h][pos][d], k[b][h][pos][d], vT[b][h][d][pos]  (f16)
// mode 1/3: outf = res + (C + bias)                                          (f32)
// mode 2: outh = gelu(C + bias)                                              (f16)
// ---------------------------------------------------------------------------
template <int K>
__global__ __launch_bounds__(256, 2)
void gemm32x64_kernel(const _Float16* __restrict__ A,
                      const _Float16* __restrict__ Bp,
                      const float* __restrict__ bias,
                      int N, int mode,
                      const float* __restrict__ res,
                      float* __restrict__ outf,
                      _Float16* __restrict__ outh,
                      _Float16* __restrict__ q_out,
                      _Float16* __restrict__ k_out,
                      _Float16* __restrict__ vT_out) {
    constexpr int KT = K >> 5;   // even, >= 32
    const int lane = threadIdx.x & 31;
    const int wave = (blockIdx.x << 3) + (threadIdx.x >> 5);
    const int nt64 = N >> 6;
    const int mt = wave / nt64;
    const int nb = wave % nt64;

    const int m    = lane & 15;
    const int koff = (lane >> 4) << 3;

    const _Float16* ap = A + ((size_t)mt * 32 + m) * K + koff;
    const _Float16* bp = Bp + ((size_t)nb * KT * 4 * 32 + lane) * 16;

    v8f acc[2][4];
#pragma unroll
    for (int h = 0; h < 2; ++h)
#pragma unroll
        for (int j = 0; j < 4; ++j)
#pragma unroll
            for (int v = 0; v < 8; ++v) acc[h][j][v] = 0.f;

    v16h a0A, a1A, bA[4];
    v16h a0B, a1B, bB[4];
    load_set<K>(ap, bp, a0A, a1A, bA);
    int kt = 0;
    for (; kt < KT - 2; kt += 2) {
        load_set<K>(ap + 32, bp + 2048, a0B, a1B, bB);
        __builtin_prefetch((const void*)(bp + 8192), 0, 0);   // stream-ahead on B
        wmma_set(a0A, a1A, bA, acc);
        ap += 64;
        bp += 4096;
        load_set<K>(ap, bp, a0A, a1A, bA);
        wmma_set(a0B, a1B, bB, acc);
    }
    load_set<K>(ap + 32, bp + 2048, a0B, a1B, bB);
    wmma_set(a0A, a1A, bA, acc);
    wmma_set(a0B, a1B, bB, acc);

    const int n0 = lane & 15;
    const int mb = (lane >> 4) << 3;
#pragma unroll
    for (int hf = 0; hf < 2; ++hf) {
#pragma unroll
        for (int j = 0; j < 4; ++j) {
            const int col = (nb << 6) + (j << 4) + n0;
            const float bcol = bias[col];
#pragma unroll
            for (int v = 0; v < 8; ++v) {
                const int row = (mt << 5) + (hf << 4) + mb + v;
                float val = acc[hf][j][v] + bcol;
                if (mode == 0) {
                    const int which = col >> 10;
                    const int d   = col & 1023;
                    const int h   = d >> 6;
                    const int dim = d & (HEAD_DIM - 1);
                    const int b_  = row >> 11;
                    const int pos = row & (L_SEQ - 1);
                    const size_t bh = (size_t)b_ * N_HEADS + h;
                    if (which == 0)
                        q_out[(bh * L_SEQ + pos) * HEAD_DIM + dim] = (_Float16)val;
                    else if (which == 1)
                        k_out[(bh * L_SEQ + pos) * HEAD_DIM + dim] = (_Float16)val;
                    else
                        vT_out[(bh * HEAD_DIM + dim) * L_SEQ + pos] = (_Float16)val;
                } else if (mode == 1) {
                    const size_t idx = (size_t)row * N + col;
                    outf[idx] = res[idx] + val;
                } else if (mode == 2) {
                    val = 0.5f * val * (1.0f + erff(val * 0.70710678118654752f));
                    outh[(size_t)row * N + col] = (_Float16)val;
                } else {
                    const size_t idx = (size_t)row * N + col;
                    outf[idx] = res[idx] + val;
                }
            }
        }
    }
}

// ---------------------------------------------------------------------------
// K-block fragment loader for attention (2 key tiles x 2 head-dim k-steps)
// ---------------------------------------------------------------------------
static __device__ __forceinline__ void load_kblock(const _Float16* Kp, int kb,
                                                   int n0, int kb16, v16h kf[4]) {
#pragma unroll
    for (int t = 0; t < 2; ++t) {
        int pos = kb + (t << 4) + n0;
        pos = pos < 0 ? 0 : (pos > L_SEQ - 1 ? L_SEQ - 1 : pos);  // OOB masked later
        const _Float16* kr = Kp + (size_t)pos * HEAD_DIM + kb16;
        kf[t * 2 + 0] = ld_frag_16(kr);
        kf[t * 2 + 1] = ld_frag_16(kr + 32);
    }
}

// ---------------------------------------------------------------------------
// Windowed flash attention. One wave per (b, h, 16-query tile); always 9
// 32-key blocks per tile. Scores via WMMA (Q pre-scaled by 1/8); tile-wide
// running max (one 5-shuffle wave reduction per block); row sums via an extra
// WMMA against an all-ones fragment. The K fragments are reloaded in place
// right after their last use, so the next block's loads overlap the whole
// softmax + PV phase with no double buffer (keeps VGPRs < budget; no spills).
// ---------------------------------------------------------------------------
__global__ __launch_bounds__(256, 2)
void attn_kernel(const _Float16* __restrict__ Q,
                 const _Float16* __restrict__ Km,
                 const _Float16* __restrict__ Vt,
                 _Float16* __restrict__ O) {
    __shared__ _Float16 plds[8][16 * 40];   // per-wave 16x32 prob tile, stride 40 (16B-aligned rows)
    const int lane = threadIdx.x & 31;
    const int wv   = threadIdx.x >> 5;
    const int wave = (blockIdx.x << 3) + wv;   // total = B*H*(L/16) = 4096

    const int qb = wave & ((L_SEQ >> 4) - 1);
    const int bh = wave >> 7;
    const int i0 = qb << 4;
    const _Float16* Qp = Q  + (size_t)bh * L_SEQ * HEAD_DIM;
    const _Float16* Kp = Km + (size_t)bh * L_SEQ * HEAD_DIM;
    const _Float16* Vp = Vt + (size_t)bh * HEAD_DIM * L_SEQ;

    const int n0   = lane & 15;
    const int hi   = lane >> 4;
    const int mb   = hi << 3;
    const int koff = hi << 4;
    const int kb16 = hi << 4;
    const int rowb = i0 + mb;            // this lane-half's first row

    // Q A-fragments, pre-scaled by 1/sqrt(HEAD_DIM) = 0.125 (exact in f16)
    v16h qa[2];
    {
        const _Float16* qr = Qp + (size_t)(i0 + n0) * HEAD_DIM;
#pragma unroll
        for (int ks = 0; ks < 2; ++ks) {
            const _Float16* p = qr + (ks << 5) + koff;
            qa[ks] = ld_frag_2x8(p, p + 8);
            qa[ks] *= (_Float16)0.125f;
        }
    }

    v16h ones;
#pragma unroll
    for (int i = 0; i < 16; ++i) ones[i] = (_Float16)1.0f;

    float gmax = -3.0e38f;
    v8f o[4], os;
#pragma unroll
    for (int v = 0; v < 8; ++v) os[v] = 0.f;
#pragma unroll
    for (int d = 0; d < 4; ++d)
#pragma unroll
        for (int v = 0; v < 8; ++v) o[d][v] = 0.f;

    const int kstart = ((i0 - WIN_HALF) >> 5) << 5;   // 32-aligned; always 9 blocks
    _Float16* pl = &plds[wv][0];

    v16h kcur[4];
    load_kblock(Kp, kstart, n0, kb16, kcur);

    for (int it = 0; it < 9; ++it) {
        const int kb = kstart + (it << 5);

        v8f s[2];
#pragma unroll
        for (int t = 0; t < 2; ++t)
#pragma unroll
            for (int v = 0; v < 8; ++v) s[t][v] = 0.f;
#pragma unroll
        for (int t = 0; t < 2; ++t)
#pragma unroll
            for (int ks = 0; ks < 2; ++ks)
                s[t] = wmma_f16(qa[ks], kcur[t * 2 + ks], s[t]);

        // prefetch next K block in place (after last use of kcur)
        if (it < 8) load_kblock(Kp, kb + 32, n0, kb16, kcur);

        // V fragments for this block (independent of probs) — overlap softmax
        v16h vb[4];
        {
            int p0i = kb + kb16;                       // 16-aligned: wholly in-range or wholly masked
            p0i = p0i < 0 ? 0 : (p0i > L_SEQ - 16 ? L_SEQ - 16 : p0i);
#pragma unroll
            for (int d = 0; d < 4; ++d)
                vb[d] = ld_frag_16(Vp + (size_t)((d << 4) + n0) * L_SEQ + p0i);
        }

        // window mask + tile-wide running max
        // col is constant over v: 1 range test per t; window test is one
        // unsigned compare: 0 <= (col - row + W) <= 2W
        float mx = -3.0e38f;
#pragma unroll
        for (int t = 0; t < 2; ++t) {
            const int col = kb + (t << 4) + n0;
            const bool colok = (col >= 0) && (col < L_SEQ);
            const int delta = col - rowb + WIN_HALF;   // inside iff (unsigned)(delta - v) <= 2W
#pragma unroll
            for (int v = 0; v < 8; ++v) {
                const bool inside = colok && ((unsigned)(delta - v) <= 2u * WIN_HALF);
                const float xx = inside ? s[t][v] : -3.0e38f;
                s[t][v] = xx;
                mx = fmaxf(mx, xx);
            }
        }
#pragma unroll
        for (int off = 1; off < 32; off <<= 1)
            mx = fmaxf(mx, __shfl_xor(mx, off, 32));
        const float nm = fmaxf(gmax, mx);
        const float sc = __expf(gmax - nm);
        gmax = nm;
#pragma unroll
        for (int v = 0; v < 8; ++v) os[v] *= sc;
#pragma unroll
        for (int d = 0; d < 4; ++d)
#pragma unroll
            for (int v = 0; v < 8; ++v) o[d][v] *= sc;

        // probabilities -> LDS (masked entries forced to exactly zero)
#pragma unroll
        for (int t = 0; t < 2; ++t)
#pragma unroll
            for (int v = 0; v < 8; ++v) {
                const float pv = (s[t][v] <= -1.0e37f) ? 0.f : __expf(s[t][v] - nm);
                pl[(mb + v) * 40 + (t << 4) + n0] = (_Float16)pv;
            }

        // probs C-layout -> A-layout via LDS, then row-sum WMMA + 4 PV WMMAs
        const _Float16* pr = pl + n0 * 40 + koff;
        const v16h pa = ld_frag_2x8(pr, pr + 8);
        os = wmma_f16(pa, ones, os);
#pragma unroll
        for (int d = 0; d < 4; ++d)
            o[d] = wmma_f16(pa, vb[d], o[d]);
    }

    const int b_ = bh >> 4;
    const int h_ = bh & 15;
#pragma unroll
    for (int v = 0; v < 8; ++v) {
        const int row = i0 + mb + v;
        const float inv = 1.0f / os[v];
        const size_t base = ((size_t)b_ * L_SEQ + row) * D_MODEL + h_ * HEAD_DIM;
#pragma unroll
        for (int d = 0; d < 4; ++d)
            O[base + (d << 4) + n0] = (_Float16)(o[d][v] * inv);
    }
}

// ---------------------------------------------------------------------------
extern "C" void kernel_launch(void* const* d_in, const int* in_sizes, int n_in,
                              void* d_out, int out_size, void* d_ws, size_t ws_size,
                              hipStream_t stream) {
    (void)in_sizes; (void)n_in; (void)out_size; (void)ws_size;
    const float* x      = (const float*)d_in[0];
    const float* qkv_w  = (const float*)d_in[1];
    const float* qkv_b  = (const float*)d_in[2];
    const float* out_w  = (const float*)d_in[3];
    const float* out_b  = (const float*)d_in[4];
    const float* ln1_g  = (const float*)d_in[5];
    const float* ln1_b  = (const float*)d_in[6];
    const float* ln2_g  = (const float*)d_in[7];
    const float* ln2_b  = (const float*)d_in[8];
    const float* ffn_w1 = (const float*)d_in[9];
    const float* ffn_b1 = (const float*)d_in[10];
    const float* ffn_w2 = (const float*)d_in[11];
    const float* ffn_b2 = (const float*)d_in[12];
    float* out = (float*)d_out;

    char* ws = (char*)d_ws;
    size_t off = 0;
    auto alloc = [&](size_t bytes) -> char* {
        char* p = ws + off;
        off += (bytes + 255) & ~(size_t)255;
        return p;
    };

    _Float16* h16    = (_Float16*)alloc((size_t)M_ROWS * D_MODEL * 2);
    _Float16* q16    = (_Float16*)alloc((size_t)M_ROWS * D_MODEL * 2);
    _Float16* k16    = (_Float16*)alloc((size_t)M_ROWS * D_MODEL * 2);
    _Float16* vT16   = (_Float16*)alloc((size_t)M_ROWS * D_MODEL * 2);
    _Float16* a16    = (_Float16*)alloc((size_t)M_ROWS * D_MODEL * 2);
    float*    x2     = (float*)   alloc((size_t)M_ROWS * D_MODEL * 4);
    _Float16* h2_16  = (_Float16*)alloc((size_t)M_ROWS * D_MODEL * 2);
    _Float16* g16    = (_Float16*)alloc((size_t)M_ROWS * 2 * D_MODEL * 2);
    _Float16* qkvw_p = (_Float16*)alloc((size_t)D_MODEL * 3 * D_MODEL * 2);
    _Float16* outw_p = (_Float16*)alloc((size_t)D_MODEL * D_MODEL * 2);
    _Float16* w1_p   = (_Float16*)alloc((size_t)D_MODEL * 2 * D_MODEL * 2);
    _Float16* w2_p   = (_Float16*)alloc((size_t)2 * D_MODEL * D_MODEL * 2);

    const dim3 blk(256);

    // weight packing into WMMA B-fragment order
    pack_w_kernel<<<(D_MODEL * 3 * D_MODEL) / 256, blk, 0, stream>>>(qkv_w, qkvw_p, D_MODEL, 3 * D_MODEL);
    pack_w_kernel<<<(D_MODEL * D_MODEL) / 256,     blk, 0, stream>>>(out_w, outw_p, D_MODEL, D_MODEL);
    pack_w_kernel<<<(D_MODEL * 2 * D_MODEL) / 256, blk, 0, stream>>>(ffn_w1, w1_p, D_MODEL, 2 * D_MODEL);
    pack_w_kernel<<<(2 * D_MODEL * D_MODEL) / 256, blk, 0, stream>>>(ffn_w2, w2_p, 2 * D_MODEL, D_MODEL);

    // LN1 -> h (f16)
    layernorm_kernel<<<M_ROWS, blk, 0, stream>>>(x, ln1_g, ln1_b, h16);

    // QKV projection, scatter to q/k/v^T
    {
        const int waves = (M_ROWS / 32) * ((3 * D_MODEL) / 64);  // 6144
        gemm32x64_kernel<D_MODEL><<<waves / 8, blk, 0, stream>>>(h16, qkvw_p, qkv_b,
            3 * D_MODEL, 0, nullptr, nullptr, nullptr, q16, k16, vT16);
    }

    // windowed attention
    attn_kernel<<<(B_SZ * N_HEADS * (L_SEQ / 16)) / 8, blk, 0, stream>>>(q16, k16, vT16, a16);

    // output projection + residual -> x2 (f32)
    {
        const int waves = (M_ROWS / 32) * (D_MODEL / 64);        // 2048
        gemm32x64_kernel<D_MODEL><<<waves / 8, blk, 0, stream>>>(a16, outw_p, out_b,
            D_MODEL, 1, x, x2, nullptr, nullptr, nullptr, nullptr);
    }

    // LN2 -> h2 (f16)
    layernorm_kernel<<<M_ROWS, blk, 0, stream>>>(x2, ln2_g, ln2_b, h2_16);

    // FFN1 + GELU -> g (f16)
    {
        const int waves = (M_ROWS / 32) * ((2 * D_MODEL) / 64);  // 4096
        gemm32x64_kernel<D_MODEL><<<waves / 8, blk, 0, stream>>>(h2_16, w1_p, ffn_b1,
            2 * D_MODEL, 2, nullptr, nullptr, g16, nullptr, nullptr, nullptr);
    }

    // FFN2 + residual -> out (f32)
    {
        const int waves = (M_ROWS / 32) * (D_MODEL / 64);        // 2048
        gemm32x64_kernel<2 * D_MODEL><<<waves / 8, blk, 0, stream>>>(g16, w2_p, ffn_b2,
            D_MODEL, 3, x2, out, nullptr, nullptr, nullptr, nullptr);
    }
}
